// MultiHeadAttentionAlibi_11690900979825
// MI455X (gfx1250) — matrix operand order repaired and checked by reading, not compiled, not run on hardware
//
#include <hip/hip_runtime.h>
#include <hip/hip_bf16.h>
#include <stdint.h>

// ---------------------------------------------------------------------------
// MultiHeadAttention + ALiBi for MI455X (gfx1250, wave32, WMMA bf16)
//   B=2, T=2048, D=1024, H=16, hd=64, slope m = 0.0625
// Strategy: fp32 -> bf16 once; all 4 GEMMs + both attention matmuls via
// v_wmma_f32_16x16x32_bf16 (f32 accum); flash-style online softmax so the
// 536MB score tensor never exists. V is produced pre-transposed so the P*V
// B-matrix is a contiguous-K load.
// ---------------------------------------------------------------------------

#define BATCH 2
#define SEQ   2048
#define DMODEL 1024
#define NHEAD 16
#define HDIM  64
#define BTROWS (BATCH * SEQ)        // 4096
#define SLOPE 0.0625f
#define SCALE 0.125f                // 1/sqrt(64)

typedef __attribute__((ext_vector_type(16))) __bf16         v16bf;
typedef __attribute__((ext_vector_type(8)))  float          v8f;
typedef __attribute__((ext_vector_type(8)))  unsigned short u16x8;
typedef __attribute__((ext_vector_type(4)))  unsigned short u16x4;

union FragU { u16x8 h[2]; v16bf v; };

__device__ __forceinline__ unsigned short f2bf(float f) {
    union { float f; unsigned u; } c; c.f = f;
    unsigned u = c.u;
    unsigned r = u + 0x7FFFu + ((u >> 16) & 1u);   // round-to-nearest-even
    return (unsigned short)(r >> 16);
}

__device__ __forceinline__ v8f wmma_bf16(v16bf a, v16bf b, v8f c) {
    return __builtin_amdgcn_wmma_f32_16x16x32_bf16(
        /*neg_a=*/false, a, /*neg_b=*/false, b,
        /*c_mod=*/(short)0, c, /*reuse_a=*/false, /*reuse_b=*/false);
}

// A-matrix 16x32 bf16 fragment: lane = row (&15);
// lo lanes hold K 0..15, hi lanes K 16..31 (matches B fragment).
// `base` = row pointer already advanced to the K base.
__device__ __forceinline__ v16bf load_a_frag(const unsigned short* base, int hi) {
    return *(const v16bf*)(base + (hi ? 16 : 0));
}

// B-matrix 32x16 bf16 fragment: lane = column; lo lanes K 0..15, hi lanes
// K 16..31 — one contiguous 32B run per lane from a K-contiguous row.
__device__ __forceinline__ v16bf load_b_frag(const unsigned short* base, int hi) {
    return *(const v16bf*)(base + (hi ? 16 : 0));
}

// ---------------------------------------------------------------------------
__global__ __launch_bounds__(256)
void f32_to_bf16_kernel(const float* __restrict__ in,
                        unsigned short* __restrict__ out, int n4) {
    int i = blockIdx.x * blockDim.x + threadIdx.x;
    if (i >= n4) return;
    float4 v = ((const float4*)in)[i];
    u16x4 o;
    o.x = f2bf(v.x); o.y = f2bf(v.y); o.z = f2bf(v.z); o.w = f2bf(v.w);
    ((u16x4*)out)[i] = o;
}

// ---------------------------------------------------------------------------
// Y = X * W^T + bias.   X: (BTROWS x D) bf16 row-major,  W: (D x D) bf16.
// Wave computes a 16x64 strip (A-fragment reused over 4 N tiles).
// mode 0: bf16 out, per-head layout  [bh][t][64]        (Q, K)
// mode 1: bf16 out, transposed head  [bh][64][t]        (V^T)
// mode 2: fp32 out, row-major        [row][D]           (final projection)
__global__ __launch_bounds__(256)
void gemm_xwt_kernel(const unsigned short* __restrict__ X,
                     const unsigned short* __restrict__ W,
                     const float* __restrict__ bias,
                     void* __restrict__ out, int mode) {
    const int lane = threadIdx.x & 31;
    const int wid  = threadIdx.x >> 5;
    const int lo   = lane & 15;
    const int hi   = lane >> 4;
    const int m0   = blockIdx.x * 128 + wid * 16;
    const int n0   = blockIdx.y * 64;

    v8f c[4] = {};
    const unsigned short* xrow = X + (size_t)(m0 + lo) * DMODEL;

    for (int k = 0; k < DMODEL; k += 32) {
        v16bf a = load_a_frag(xrow + k, hi);
#pragma unroll
        for (int nt = 0; nt < 4; ++nt) {
            const unsigned short* wr =
                W + (size_t)(n0 + nt * 16 + lo) * DMODEL + k;
            v16bf b = load_b_frag(wr, hi);
            c[nt] = wmma_bf16(a, b, c[nt]);
        }
    }

#pragma unroll
    for (int nt = 0; nt < 4; ++nt) {
        const int col = n0 + nt * 16 + lo;
        const float bc = bias[col];
#pragma unroll
        for (int r = 0; r < 8; ++r) {
            const int row = m0 + r + hi * 8;          // global (b*T + t)
            const float v = c[nt][r] + bc;
            if (mode == 2) {
                ((float*)out)[(size_t)row * DMODEL + col] = v;
            } else {
                const int bb = row >> 11;              // / SEQ
                const int t  = row & (SEQ - 1);
                const int h  = col >> 6;
                const int d  = col & (HDIM - 1);
                const unsigned short bv = f2bf(v);
                unsigned short* o = (unsigned short*)out;
                if (mode == 0)
                    o[((size_t)(bb * NHEAD + h) * SEQ + t) * HDIM + d] = bv;
                else
                    o[((size_t)(bb * NHEAD + h) * HDIM + d) * SEQ + t] = bv;
            }
        }
    }
}

// ---------------------------------------------------------------------------
// Flash attention, causal + ALiBi.  Block = 8 waves x 16 Q-rows = 128 rows of
// one (b,h). K streamed in 64-wide blocks; P repacked C-layout -> A-layout
// through a private LDS tile per wave.
__global__ __launch_bounds__(256)
void flash_attn_kernel(const unsigned short* __restrict__ Qh,
                       const unsigned short* __restrict__ Kh,
                       const unsigned short* __restrict__ Vt,   // [bh][64][T]
                       unsigned short* __restrict__ Ob) {       // [b*T][D] bf16
    __shared__ unsigned short pbuf[8][16][64];                  // 16 KB

    const int lane = threadIdx.x & 31;
    const int wid  = threadIdx.x >> 5;
    const int lo   = lane & 15;
    const int hi   = lane >> 4;
    const int bh   = blockIdx.z * NHEAD + blockIdx.y;
    const int q0   = blockIdx.x * 128;
    const int qm   = q0 + wid * 16;

    const unsigned short* Q = Qh + (size_t)bh * SEQ * HDIM;
    const unsigned short* K = Kh + (size_t)bh * SEQ * HDIM;
    const unsigned short* V = Vt + (size_t)bh * HDIM * SEQ;

    // Q fragments for this wave's 16 rows (hd=64 -> two K=32 steps), persistent
    const unsigned short* qrow = Q + (size_t)(qm + lo) * HDIM;
    const v16bf aq0 = load_a_frag(qrow, hi);
    const v16bf aq1 = load_a_frag(qrow + 32, hi);

    v8f o[4] = {};
    v8f lrow = {};
    v8f mrow;
#pragma unroll
    for (int r = 0; r < 8; ++r) mrow[r] = -1e30f;

    const int nkb = (q0 + 128) / 64;   // causal: K blocks covering k <= q0+127
    for (int kb = 0; kb < nkb; ++kb) {
        const int kt0 = kb * 64;

        // ---- S = Q * K^T  (4 tiles x 2 k-steps) ----
        v8f s[4];
#pragma unroll
        for (int nt = 0; nt < 4; ++nt) {
            const unsigned short* kr = K + (size_t)(kt0 + nt * 16 + lo) * HDIM;
            v8f acc = {};
            acc = wmma_bf16(aq0, load_b_frag(kr, hi), acc);
            acc = wmma_bf16(aq1, load_b_frag(kr + 32, hi), acc);
            s[nt] = acc;
        }
        if (kb + 1 < nkb)  // warm L2/WGP$ for next K block
            __builtin_prefetch(K + (size_t)(kt0 + 64 + lo) * HDIM, 0, 3);

        // ---- scale + ALiBi + causal mask ----
#pragma unroll
        for (int nt = 0; nt < 4; ++nt) {
            const int kg = kt0 + nt * 16 + lo;
#pragma unroll
            for (int r = 0; r < 8; ++r) {
                const int qg = qm + r + hi * 8;
                const int dlt = kg - qg;
                float v = s[nt][r] * SCALE + SLOPE * (float)dlt;
                s[nt][r] = (dlt <= 0) ? v : -1e30f;
            }
        }

        // ---- online softmax: row max over 64 cols ----
        v8f mnew = mrow;
#pragma unroll
        for (int nt = 0; nt < 4; ++nt)
#pragma unroll
            for (int r = 0; r < 8; ++r) mnew[r] = fmaxf(mnew[r], s[nt][r]);
#pragma unroll
        for (int msk = 1; msk <= 8; msk <<= 1)
#pragma unroll
            for (int r = 0; r < 8; ++r)
                mnew[r] = fmaxf(mnew[r], __shfl_xor(mnew[r], msk, 32));

        v8f alpha;
#pragma unroll
        for (int r = 0; r < 8; ++r) alpha[r] = __expf(mrow[r] - mnew[r]);
        mrow = mnew;
#pragma unroll
        for (int nt = 0; nt < 4; ++nt)
#pragma unroll
            for (int r = 0; r < 8; ++r) o[nt][r] *= alpha[r];

        // ---- P = exp(S - m); row-sum; stash bf16 P in per-wave LDS tile ----
        v8f psum = {};
#pragma unroll
        for (int nt = 0; nt < 4; ++nt)
#pragma unroll
            for (int r = 0; r < 8; ++r) {
                const float sv = s[nt][r];
                const float p = (sv > -5e29f) ? __expf(sv - mnew[r]) : 0.0f;
                psum[r] += p;
                pbuf[wid][r + hi * 8][nt * 16 + lo] = f2bf(p);
            }
#pragma unroll
        for (int msk = 1; msk <= 8; msk <<= 1)
#pragma unroll
            for (int r = 0; r < 8; ++r) psum[r] += __shfl_xor(psum[r], msk, 32);
#pragma unroll
        for (int r = 0; r < 8; ++r) lrow[r] = lrow[r] * alpha[r] + psum[r];

        // ---- O += P * V   (A from LDS repack, B from transposed V) ----
        const v16bf ap0 = load_a_frag(&pbuf[wid][lo][0], hi);
        const v16bf ap1 = load_a_frag(&pbuf[wid][lo][32], hi);
#pragma unroll
        for (int nt = 0; nt < 4; ++nt) {
            const unsigned short* vr = V + (size_t)(nt * 16 + lo) * SEQ + kt0;
            o[nt] = wmma_bf16(ap0, load_b_frag(vr, hi), o[nt]);
            o[nt] = wmma_bf16(ap1, load_b_frag(vr + 32, hi), o[nt]);
        }
    }

    // ---- finalize: O /= l, store bf16 in [b*T][D] at columns h*64.. ----
    v8f rinv;
#pragma unroll
    for (int r = 0; r < 8; ++r) rinv[r] = 1.0f / lrow[r];
    const int colbase = blockIdx.y * HDIM;
#pragma unroll
    for (int nt = 0; nt < 4; ++nt)
#pragma unroll
        for (int r = 0; r < 8; ++r) {
            const int qrw = qm + r + hi * 8;
            Ob[((size_t)(blockIdx.z * SEQ + qrw)) * DMODEL +
               colbase + nt * 16 + lo] = f2bf(o[nt][r] * rinv[r]);
        }
}

// ---------------------------------------------------------------------------
extern "C" void kernel_launch(void* const* d_in, const int* in_sizes, int n_in,
                              void* d_out, int out_size, void* d_ws, size_t ws_size,
                              hipStream_t stream) {
    (void)in_sizes; (void)n_in; (void)out_size; (void)ws_size;
    const float* x  = (const float*)d_in[0];
    const float* Wq = (const float*)d_in[1];
    const float* bq = (const float*)d_in[2];
    const float* Wk = (const float*)d_in[3];
    const float* bk = (const float*)d_in[4];
    const float* Wv = (const float*)d_in[5];
    const float* bv = (const float*)d_in[6];
    const float* Wo = (const float*)d_in[7];
    const float* bo = (const float*)d_in[8];

    char* ws = (char*)d_ws;                       // needs ~48 MB
    unsigned short* xb  = (unsigned short*)(ws);                        //  8 MB
    unsigned short* wqb = (unsigned short*)(ws + ((size_t) 8 << 20));   //  2 MB
    unsigned short* wkb = (unsigned short*)(ws + ((size_t)10 << 20));
    unsigned short* wvb = (unsigned short*)(ws + ((size_t)12 << 20));
    unsigned short* wob = (unsigned short*)(ws + ((size_t)14 << 20));
    unsigned short* qh  = (unsigned short*)(ws + ((size_t)16 << 20));   //  8 MB
    unsigned short* kh  = (unsigned short*)(ws + ((size_t)24 << 20));   //  8 MB
    unsigned short* vt  = (unsigned short*)(ws + ((size_t)32 << 20));   //  8 MB
    unsigned short* ob  = (unsigned short*)(ws + ((size_t)40 << 20));   //  8 MB

    const int nx = BTROWS * DMODEL;     // 4,194,304
    const int nw = DMODEL * DMODEL;     // 1,048,576
    f32_to_bf16_kernel<<<nx / 1024, 256, 0, stream>>>(x,  xb,  nx / 4);
    f32_to_bf16_kernel<<<nw / 1024, 256, 0, stream>>>(Wq, wqb, nw / 4);
    f32_to_bf16_kernel<<<nw / 1024, 256, 0, stream>>>(Wk, wkb, nw / 4);
    f32_to_bf16_kernel<<<nw / 1024, 256, 0, stream>>>(Wv, wvb, nw / 4);
    f32_to_bf16_kernel<<<nw / 1024, 256, 0, stream>>>(Wo, wob, nw / 4);

    dim3 gg(BTROWS / 128, DMODEL / 64);           // (32, 16)
    gemm_xwt_kernel<<<gg, 256, 0, stream>>>(xb, wqb, bq, qh, 0);
    gemm_xwt_kernel<<<gg, 256, 0, stream>>>(xb, wkb, bk, kh, 0);
    gemm_xwt_kernel<<<gg, 256, 0, stream>>>(xb, wvb, bv, vt, 1);

    flash_attn_kernel<<<dim3(SEQ / 128, NHEAD, BATCH), 256, 0, stream>>>(
        qh, kh, vt, ob);

    gemm_xwt_kernel<<<gg, 256, 0, stream>>>(ob, wob, bo, d_out, 2);
}
